// DeepSeekAttention_63934883169033
// MI455X (gfx1250) — matrix operand order repaired and checked by reading, not compiled, not run on hardware
//
#include <hip/hip_runtime.h>
#include <hip/hip_bf16.h>

// ---------------------------------------------------------------------------
// Problem constants (match reference)
// ---------------------------------------------------------------------------
static constexpr int B_  = 2;
static constexpr int S_  = 2048;
static constexpr int H_  = 2048;
static constexpr int NH  = 16;
static constexpr int NKV = 4;
static constexpr int HD  = 128;
static constexpr int BS  = B_ * S_;          // 4096 rows

typedef __bf16 bf16_t;
typedef __attribute__((ext_vector_type(16))) __bf16 v16bf;
typedef __attribute__((ext_vector_type(8)))  float  v8f;

#if __has_builtin(__builtin_amdgcn_tensor_load_to_lds) && \
    __has_builtin(__builtin_amdgcn_s_wait_tensorcnt)
#define HAVE_TDM 1
typedef __attribute__((ext_vector_type(4))) unsigned tdm_u32x4;
typedef __attribute__((ext_vector_type(8))) int      tdm_i32x8;
typedef __attribute__((ext_vector_type(4))) int      tdm_i32x4;
#else
#define HAVE_TDM 0
#endif

union FragBF { v16bf v; unsigned u[8]; };

// ---------------------------------------------------------------------------
// WMMA fragment loaders (layouts per CDNA5 ISA §7.12.2, wave32)
// ---------------------------------------------------------------------------
__device__ inline v16bf load_frag_a(const bf16_t* base, int strideElems, int lane) {
  const int m = lane & 15, h = lane >> 4;
  const bf16_t* row = base + m * strideElems;
  FragBF f;
#pragma unroll
  for (int j = 0; j < 4; ++j) f.u[j]     = *(const unsigned*)(row + h * 8 + 2 * j);
#pragma unroll
  for (int j = 0; j < 4; ++j) f.u[4 + j] = *(const unsigned*)(row + 16 + h * 8 + 2 * j);
  return f.v;
}

__device__ inline v16bf load_frag_b(const bf16_t* base, int strideElems, int lane) {
  const int n = lane & 15, h = lane >> 4;
  const bf16_t* row = base + n * strideElems + h * 8;
  FragBF f;
#pragma unroll
  for (int j = 0; j < 8; ++j) f.u[j] = *(const unsigned*)(row + (j >> 2) * 16 + 2 * (j & 3));
  return f.v;
}

__device__ inline v8f zero8() { v8f z = {}; return z; }

// ---------------------------------------------------------------------------
// fp32 -> bf16 cast
// ---------------------------------------------------------------------------
__global__ void cast_f32_bf16_kernel(const float* __restrict__ in,
                                     bf16_t* __restrict__ out, int n) {
  int i = blockIdx.x * blockDim.x + threadIdx.x;
  if (i < n) out[i] = (bf16_t)in[i];
}

// ---------------------------------------------------------------------------
// GEMM: C[M,N] (f32) = A[M,K] (bf16) * Bw[N,K]^T (bf16)   (nn.Linear layout)
// Block 128x128, K-step 64, 8 waves (4x2), wave tile 32x64.
// Software pipelined: global->regs for step k+1 overlaps WMMA on step k.
// ---------------------------------------------------------------------------
template <int N>
__global__ __launch_bounds__(256) void gemm_bf16_nt_kernel(
    const bf16_t* __restrict__ A, const bf16_t* __restrict__ Bw,
    float* __restrict__ C, int M, int K) {
  __shared__ bf16_t As[128][72];   // 144B row stride: 16B multiple, conflict-free
  __shared__ bf16_t Bs[128][72];

  const int bm = blockIdx.x * 128;
  const int bn = blockIdx.y * 128;
  const int t = threadIdx.x;
  const int lane = t & 31, wave = t >> 5;
  const int wm = (wave & 3) * 32;
  const int wn = (wave >> 2) * 64;

  v8f acc[2][4];
#pragma unroll
  for (int i = 0; i < 2; ++i)
#pragma unroll
    for (int j = 0; j < 4; ++j) acc[i][j] = zero8();

  uint4 ra[4], rb[4];   // register staging for the 128x64 bf16 tiles

  auto gload = [&](int k0) {
#pragma unroll
    for (int c = 0; c < 4; ++c) {
      int e = (c * 256 + t) * 8;
      int row = e >> 6, col = e & 63;
      ra[c] = *(const uint4*)&A[(size_t)(bm + row) * K + k0 + col];
      rb[c] = *(const uint4*)&Bw[(size_t)(bn + row) * K + k0 + col];
    }
  };
  auto lstore = [&]() {
#pragma unroll
    for (int c = 0; c < 4; ++c) {
      int e = (c * 256 + t) * 8;
      int row = e >> 6, col = e & 63;
      *(uint4*)&As[row][col] = ra[c];
      *(uint4*)&Bs[row][col] = rb[c];
    }
  };

  gload(0);
  lstore();
  for (int k0 = 64; k0 <= K; k0 += 64) {
    __syncthreads();                 // LDS tile (k0-64) ready
    if (k0 < K) gload(k0);           // prefetch next tile into registers
#pragma unroll
    for (int kc = 0; kc < 2; ++kc) { // two 32-K WMMA chunks per staged tile
      v16bf af[2], bfv[4];
#pragma unroll
      for (int i = 0; i < 2; ++i) af[i]  = load_frag_a(&As[wm + i * 16][kc * 32], 72, lane);
#pragma unroll
      for (int j = 0; j < 4; ++j) bfv[j] = load_frag_b(&Bs[wn + j * 16][kc * 32], 72, lane);
#pragma unroll
      for (int i = 0; i < 2; ++i)
#pragma unroll
        for (int j = 0; j < 4; ++j)
          acc[i][j] = __builtin_amdgcn_wmma_f32_16x16x32_bf16(
              false, af[i], false, bfv[j], (short)0, acc[i][j], false, false);
    }
    __syncthreads();                 // everyone done reading LDS
    if (k0 < K) lstore();            // commit next tile
  }

  // epilogue: one base pointer per lane, compile-time store offsets
  const int n0 = lane & 15, halfm = lane >> 4;
  float* cbase = C + (size_t)(bm + wm + halfm * 8) * N + (bn + wn + n0);
#pragma unroll
  for (int i = 0; i < 2; ++i)
#pragma unroll
    for (int r = 0; r < 8; ++r) {
      float* p = cbase + (size_t)(i * 16 + r) * N;
#pragma unroll
      for (int j = 0; j < 4; ++j) p[j * 16] = acc[i][j][r];
    }
}

// ---------------------------------------------------------------------------
// RoPE + cast + head-transpose: X[BS, nh*HD] f32 -> Out[B, nh, S, HD] bf16
// ---------------------------------------------------------------------------
__global__ void rope_cast_kernel(const float* __restrict__ X,
                                 const float* __restrict__ cosb,
                                 const float* __restrict__ sinb,
                                 bf16_t* __restrict__ Out, int nh, int do_rope) {
  int i = blockIdx.x * blockDim.x + threadIdx.x;   // over B*S*nh*64
  int total = BS * nh * 64;
  if (i >= total) return;
  int d2 = i & 63;
  int rest = i >> 6;
  int h = rest % nh;
  int bs = rest / nh;
  int s = bs & (S_ - 1);
  int b = bs >> 11;

  const float* xr = X + (size_t)bs * (nh * HD) + h * HD;
  float x1 = xr[d2], x2 = xr[d2 + 64];
  float o1, o2;
  if (do_rope) {
    float c1 = cosb[s * HD + d2],      s1 = sinb[s * HD + d2];
    float c2 = cosb[s * HD + d2 + 64], s2 = sinb[s * HD + d2 + 64];
    o1 = x1 * c1 - x2 * s1;
    o2 = x2 * c2 + x1 * s2;
  } else {
    o1 = x1; o2 = x2;
  }
  bf16_t* orow = Out + (((size_t)b * nh + h) * S_ + s) * HD;
  orow[d2]      = (bf16_t)o1;
  orow[d2 + 64] = (bf16_t)o2;
}

// ---------------------------------------------------------------------------
// Flash attention, causal, GQA (head h -> kv head h/NKV).
// Grid: (S/128, NH, B). 8 waves x 16 queries. 64-key tiles.
// K tile staged by the Tensor Data Mover (wave 0) when available; V tile
// transposed manually (TDM cannot transpose).
// ---------------------------------------------------------------------------
__global__ __launch_bounds__(256) void flash_attn_kernel(
    const bf16_t* __restrict__ Q, const bf16_t* __restrict__ K,
    const bf16_t* __restrict__ V, bf16_t* __restrict__ O) {
  __shared__ bf16_t Ks[64][136];      // [key][d]  row = 64 DW data + 4 DW pad
  __shared__ bf16_t Vt[128][72];      // [d][key]
  __shared__ bf16_t Ps[8][16][72];    // per-wave P relayout buffer

  const int qblk = blockIdx.x;
  const int h    = blockIdx.y;
  const int b    = blockIdx.z;
  const int kvh  = h / NKV;

  const int t = threadIdx.x, lane = t & 31, wave = t >> 5;
  const int laneN = lane & 15, laneH = lane >> 4;
  const int qw = qblk * 128 + wave * 16;

  const bf16_t* Qbase = Q + ((size_t)b * NH + h) * S_ * HD;
  const bf16_t* Kbase = K + ((size_t)b * NKV + kvh) * S_ * HD;
  const bf16_t* Vbase = V + ((size_t)b * NKV + kvh) * S_ * HD;

  // Q fragments (A layout), resident in registers for the whole kernel
  v16bf qf[4];
  {
    const bf16_t* qrow = Qbase + (size_t)(qw + laneN) * HD;
#pragma unroll
    for (int ks = 0; ks < 4; ++ks) {
      FragBF f;
#pragma unroll
      for (int j = 0; j < 4; ++j)
        f.u[j] = *(const unsigned*)(qrow + ks * 32 + laneH * 8 + 2 * j);
#pragma unroll
      for (int j = 0; j < 4; ++j)
        f.u[4 + j] = *(const unsigned*)(qrow + ks * 32 + 16 + laneH * 8 + 2 * j);
      qf[ks] = f.v;
    }
  }

#if HAVE_TDM
  // TDM descriptor (CDNA5 ISA): 2-D tile 128x64 bf16,
  // LDS padding 4 DW after every 64 DW -> reproduces Ks[64][136].
  tdm_u32x4 g0;
  tdm_i32x8 g1;
  tdm_i32x4 g2 = {}, g3 = {};
  tdm_i32x8 g4 = {};                   // groups 2/3 extension words (unused, 2-D)
  {
    unsigned lds = (unsigned)(size_t)&Ks[0][0];
    g0.x = 1u;                         // count=1, user mode, no gather
    g0.y = lds;                        // lds_addr
    g0.z = 0u;                         // global_addr low (per tile, set in loop)
    g0.w = 0u;
    g1[0] = (1 << 16)                  // data_size = 2B
          | (1 << 20)                  // pad_enable
          | (5 << 22)                  // pad_interval: 64 DWORDs
          | (3 << 25);                 // pad_amount: 4 DWORDs
    g1[1] = (HD << 16);                // tensor_dim0 = 128 (bits 79:48)
    g1[2] = (S_ & 0xFFFF) << 16;       // tensor_dim1 low (bits 111:80)
    g1[3] = (HD << 16);                // tile_dim0 = 128 (bits 127:112)
    g1[4] = 64;                        // tile_dim1 = 64 keys
    g1[5] = HD;                        // tensor_dim0_stride = 128 elements
    g1[6] = 0;
    g1[7] = 0;
  }
#endif

  v8f oacc[8];
#pragma unroll
  for (int i = 0; i < 8; ++i) oacc[i] = zero8();
  float mrow[8], lrow[8];
#pragma unroll
  for (int r = 0; r < 8; ++r) { mrow[r] = -1e30f; lrow[r] = 0.f; }
  const float scale = 0.08838834764831845f;   // 1/sqrt(128)

  const int nkt = 2 * qblk + 2;               // causal
  for (int kt = 0; kt < nkt; ++kt) {
    const int k0 = kt * 64;
    __syncthreads();                          // previous tile fully consumed

#if HAVE_TDM
    if (wave == 0) {                          // per-wave op; one wave issues
      unsigned long long ga = (unsigned long long)(size_t)(Kbase + (size_t)k0 * HD);
      g0.z = (unsigned)ga;
      g0.w = (unsigned)(ga >> 32) | (2u << 30);   // type=2 ("image")
      __builtin_amdgcn_tensor_load_to_lds(g0, g1, g2, g3, g4, 0);
    }
#endif
    // V tile: load rows, transpose into Vt (all waves, overlaps TDM)
#pragma unroll
    for (int c = 0; c < 4; ++c) {
      int e = (c * 256 + t) * 8;
      int row = e >> 7, col = e & 127;
#if !HAVE_TDM
      *(uint4*)&Ks[row][col] = *(const uint4*)&Kbase[(size_t)(k0 + row) * HD + col];
#endif
      uint4 vv = *(const uint4*)&Vbase[(size_t)(k0 + row) * HD + col];
      const bf16_t* pv = (const bf16_t*)&vv;
#pragma unroll
      for (int x = 0; x < 8; ++x) Vt[col + x][row] = pv[x];
    }
#if HAVE_TDM
    if (wave == 0) __builtin_amdgcn_s_wait_tensorcnt(0);
#endif
    __syncthreads();

    // ---- scores: S[16q x 64k] = Q * K^T ----
    // ks-outer ordering: each batch = 4 independent B-frag loads feeding 4
    // independent WMMAs (distinct accumulators), so LDS loads of batch ks+1
    // overlap the WMMAs of batch ks instead of serializing on s_wait_dscnt.
    v8f a4[4];
#pragma unroll
    for (int ns = 0; ns < 4; ++ns) a4[ns] = zero8();
#pragma unroll
    for (int ks = 0; ks < 4; ++ks) {
      v16bf bk[4];
#pragma unroll
      for (int ns = 0; ns < 4; ++ns)
        bk[ns] = load_frag_b(&Ks[ns * 16][ks * 32], 136, lane);
#pragma unroll
      for (int ns = 0; ns < 4; ++ns)
        a4[ns] = __builtin_amdgcn_wmma_f32_16x16x32_bf16(
            false, qf[ks], false, bk[ns], (short)0, a4[ns], false, false);
    }
    float sc[4][8];
#pragma unroll
    for (int ns = 0; ns < 4; ++ns) {
      int key = k0 + ns * 16 + laneN;
#pragma unroll
      for (int r = 0; r < 8; ++r) {
        int qrow = qw + laneH * 8 + r;
        float s = a4[ns][r] * scale;
        sc[ns][r] = (key <= qrow) ? s : -1e30f;
      }
    }

    // ---- online softmax ----
#pragma unroll
    for (int r = 0; r < 8; ++r) {
      float mx = fmaxf(fmaxf(sc[0][r], sc[1][r]), fmaxf(sc[2][r], sc[3][r]));
      mx = fmaxf(mx, __shfl_xor(mx, 1, 16));
      mx = fmaxf(mx, __shfl_xor(mx, 2, 16));
      mx = fmaxf(mx, __shfl_xor(mx, 4, 16));
      mx = fmaxf(mx, __shfl_xor(mx, 8, 16));
      float mnew  = fmaxf(mrow[r], mx);
      float alpha = __expf(mrow[r] - mnew);
      float lsum = 0.f;
#pragma unroll
      for (int ns = 0; ns < 4; ++ns) {
        float p = __expf(sc[ns][r] - mnew);
        sc[ns][r] = p;
        lsum += p;
      }
      lsum += __shfl_xor(lsum, 1, 16);
      lsum += __shfl_xor(lsum, 2, 16);
      lsum += __shfl_xor(lsum, 4, 16);
      lsum += __shfl_xor(lsum, 8, 16);
      lrow[r] = lrow[r] * alpha + lsum;
      mrow[r] = mnew;
#pragma unroll
      for (int os = 0; os < 8; ++os) oacc[os][r] *= alpha;
    }

    // ---- P: C-layout regs -> per-wave LDS (A-layout source) ----
#pragma unroll
    for (int ns = 0; ns < 4; ++ns)
#pragma unroll
      for (int r = 0; r < 8; ++r)
        Ps[wave][laneH * 8 + r][ns * 16 + laneN] = (bf16_t)sc[ns][r];

    // ---- O += P * V ----  (independent per-os accumulator chains)
#pragma unroll
    for (int kc = 0; kc < 2; ++kc) {
      v16bf pa = load_frag_a(&Ps[wave][0][kc * 32], 72, lane);
#pragma unroll
      for (int os = 0; os < 8; ++os) {
        v16bf vb = load_frag_b(&Vt[os * 16][kc * 32], 72, lane);
        oacc[os] = __builtin_amdgcn_wmma_f32_16x16x32_bf16(
            false, pa, false, vb, (short)0, oacc[os], false, false);
      }
    }
  }

  // epilogue: normalize, write bf16 to [B*S, NH*HD]
  bf16_t* obase = O + ((size_t)(b * S_ + qw + laneH * 8)) * (NH * HD) + h * HD + laneN;
#pragma unroll
  for (int r = 0; r < 8; ++r) {
    float inv = 1.f / lrow[r];
    bf16_t* orow = obase + (size_t)r * (NH * HD);
#pragma unroll
    for (int os = 0; os < 8; ++os)
      orow[os * 16] = (bf16_t)(oacc[os][r] * inv);
  }
}

// ---------------------------------------------------------------------------
// Host launcher
// ---------------------------------------------------------------------------
extern "C" void kernel_launch(void* const* d_in, const int* in_sizes, int n_in,
                              void* d_out, int out_size, void* d_ws, size_t ws_size,
                              hipStream_t stream) {
  const float* x    = (const float*)d_in[0];
  const float* Wq   = (const float*)d_in[1];
  const float* Wk   = (const float*)d_in[2];
  const float* Wv   = (const float*)d_in[3];
  const float* Wo   = (const float*)d_in[4];
  const float* cosb = (const float*)d_in[5];
  const float* sinb = (const float*)d_in[6];
  // d_in[7] = attention_mask: exactly causal -> applied analytically.

  char* ws = (char*)d_ws;
  size_t off = 0;
  auto alloc = [&](size_t bytes) -> char* {
    char* p = ws + off;
    off += (bytes + 255) & ~(size_t)255;
    return p;
  };

  bf16_t* Xbf  = (bf16_t*)alloc((size_t)BS * H_ * 2);
  bf16_t* Wqb  = (bf16_t*)alloc((size_t)NH * HD * H_ * 2);
  bf16_t* Wkb  = (bf16_t*)alloc((size_t)NKV * HD * H_ * 2);
  bf16_t* Wvb  = (bf16_t*)alloc((size_t)NKV * HD * H_ * 2);
  bf16_t* Wob  = (bf16_t*)alloc((size_t)H_ * NH * HD * 2);
  float*  Qf   = (float*)alloc((size_t)BS * NH * HD * 4);
  float*  Kf   = (float*)alloc((size_t)BS * NKV * HD * 4);
  float*  Vf   = (float*)alloc((size_t)BS * NKV * HD * 4);
  bf16_t* Qbf  = (bf16_t*)alloc((size_t)BS * NH * HD * 2);
  bf16_t* Kbf  = (bf16_t*)alloc((size_t)BS * NKV * HD * 2);
  bf16_t* Vbf  = (bf16_t*)alloc((size_t)BS * NKV * HD * 2);
  bf16_t* Abf  = (bf16_t*)Qf;   // attn output aliases Qf (dead by then)

  auto cast = [&](const float* src, bf16_t* dst, int n) {
    cast_f32_bf16_kernel<<<(n + 255) / 256, 256, 0, stream>>>(src, dst, n);
  };
  cast(x,  Xbf, BS * H_);
  cast(Wq, Wqb, NH * HD * H_);
  cast(Wk, Wkb, NKV * HD * H_);
  cast(Wv, Wvb, NKV * HD * H_);
  cast(Wo, Wob, H_ * NH * HD);

  gemm_bf16_nt_kernel<2048><<<dim3(BS / 128, (NH * HD) / 128), 256, 0, stream>>>(
      Xbf, Wqb, Qf, BS, H_);
  gemm_bf16_nt_kernel<512><<<dim3(BS / 128, (NKV * HD) / 128), 256, 0, stream>>>(
      Xbf, Wkb, Kf, BS, H_);
  gemm_bf16_nt_kernel<512><<<dim3(BS / 128, (NKV * HD) / 128), 256, 0, stream>>>(
      Xbf, Wvb, Vf, BS, H_);

  {
    int nq = BS * NH * 64, nk = BS * NKV * 64;
    rope_cast_kernel<<<(nq + 255) / 256, 256, 0, stream>>>(Qf, cosb, sinb, Qbf, NH, 1);
    rope_cast_kernel<<<(nk + 255) / 256, 256, 0, stream>>>(Kf, cosb, sinb, Kbf, NKV, 1);
    rope_cast_kernel<<<(nk + 255) / 256, 256, 0, stream>>>(Vf, cosb, sinb, Vbf, NKV, 0);
  }

  flash_attn_kernel<<<dim3(S_ / 128, NH, B_), 256, 0, stream>>>(Qbf, Kbf, Vbf, Abf);

  gemm_bf16_nt_kernel<2048><<<dim3(BS / 128, H_ / 128), 256, 0, stream>>>(
      Abf, Wob, (float*)d_out, BS, NH * HD);

  (void)in_sizes; (void)n_in; (void)out_size; (void)ws_size;
}
